// NMT_32779190403732
// MI455X (gfx1250) — matrix operand order repaired and checked by reading, not compiled, not run on hardware
//
#include <hip/hip_runtime.h>
#include <cstdint>

typedef _Float16 h16;
typedef __attribute__((ext_vector_type(16))) _Float16 v16h;
typedef __attribute__((ext_vector_type(8)))  float    v8f;
typedef __attribute__((ext_vector_type(4)))  unsigned int v4u;
typedef __attribute__((ext_vector_type(8)))  int      v8i;
typedef __attribute__((ext_vector_type(4)))  int      v4i;

struct Seg  { const h16* A; long as; const h16* W; long ws; };
struct GemmArgs { Seg s[3]; int nseg; int K; };

__device__ __forceinline__ float sigm(float x) { return 1.f / (1.f + __expf(-x)); }

// ---- WMMA fragment helpers (ISA 7.12.2 layouts, wave32) ----
__device__ __forceinline__ v16h load_a_frag(const h16* A, long rs) {
  int lane = threadIdx.x & 31;
  int row = lane & 15, hi = lane >> 4;
  const h16* p = A + (long)row * rs;
  v16h a;
#pragma unroll
  for (int v = 0; v < 8; ++v) {
    int k0 = ((v & 3) << 1) + (hi << 3) + ((v >> 2) << 4);
    a[2 * v]     = p[k0];
    a[2 * v + 1] = p[k0 + 1];
  }
  return a;
}

__device__ __forceinline__ v16h load_b_frag(const h16* W, long rs) {
  int lane = threadIdx.x & 31;
  int col = lane & 15, hi = lane >> 4;
  const h16* p = W + (long)col * rs;
  v16h b;
#pragma unroll
  for (int e = 0; e < 16; ++e) b[e] = p[(e & 7) + (hi << 3) + ((e >> 3) << 4)];
  return b;
}

__device__ __forceinline__ v8f wmma_f16(v16h a, v16h b, v8f c) {
  return __builtin_amdgcn_wmma_f32_16x16x32_f16(false, a, false, b, (short)0, c,
                                                false, false);
}

// ---- elementwise utility kernels ----
__global__ void k_f32_to_f16(const float* __restrict__ src, h16* __restrict__ dst, long n) {
  long i = (long)blockIdx.x * 256 + threadIdx.x;
  long stride = (long)gridDim.x * 256;
  for (; i < n; i += stride) dst[i] = (h16)src[i];
}

__global__ void k_gather16(const float* __restrict__ emb, const int* __restrict__ tok,
                           h16* __restrict__ dst) {
  int row = blockIdx.x;                       // t*32+b, 1536 rows
  long s = (long)tok[row] * 512;
  for (int e = threadIdx.x; e < 512; e += 256)
    dst[(long)row * 512 + e] = (h16)emb[s + e];
}

__global__ void k_pack16(const float* __restrict__ src, h16* __restrict__ dst,
                         long rs, long coloff) {
  int i = blockIdx.x * 256 + threadIdx.x;     // 32*512 elements
  int b = i >> 9, c = i & 511;
  dst[(long)b * rs + coloff + c] = (h16)src[i];
}

// ---- generic tiled GEMM: C[M,N] = act( sum_seg A16 @ W16^T + bias ) ----
// one wave computes a 16x64 tile (1 A-frag reused by 4 WMMAs per K-step)
__global__ void k_gemm(GemmArgs ga, const float* __restrict__ bias,
                       float* __restrict__ outF, h16* __restrict__ outH,
                       long out_rs, int do_tanh) {
  int mtile = blockIdx.x, ng = blockIdx.y;     // ng covers 4 consecutive N-tiles
  int lane = threadIdx.x & 31;
  v8f acc[4] = {};
  for (int kb = 0; kb < ga.K; kb += 32) {
    const Seg& sg = ga.s[kb >> 9];
    int kl = kb & 511;
    v16h a = load_a_frag(sg.A + (long)mtile * 16 * sg.as + kl, sg.as);
#pragma unroll
    for (int j = 0; j < 4; ++j) {
      v16h b = load_b_frag(sg.W + (long)(ng * 4 + j) * 16 * sg.ws + kl, sg.ws);
      acc[j] = wmma_f16(a, b, acc[j]);
    }
  }
  int col = lane & 15, hi = lane >> 4;
#pragma unroll
  for (int j = 0; j < 4; ++j) {
    int n = (ng * 4 + j) * 16 + col;
    float bv = bias ? bias[n] : 0.f;
#pragma unroll
    for (int r = 0; r < 8; ++r) {
      int m = mtile * 16 + r + 8 * hi;
      float v = acc[j][r] + bv;
      if (do_tanh) v = tanhf(v);
      if (outF) outF[(long)m * out_rs + n] = v;
      if (outH) outH[(long)m * out_rs + n] = (h16)v;
    }
  }
}

// ---- fused LSTM cell step: gates via WMMA, cell math, h16 for next step ----
// grid 32 blocks (16 h-columns each), 128 threads (4 waves = 4 gates)
__global__ void k_lstm_step(GemmArgs ga, const float* __restrict__ pre,
                            const float* __restrict__ bias,
                            float* __restrict__ h32, float* __restrict__ c32,
                            h16* __restrict__ h16out, long h16rs) {
  __shared__ float g[4][32][16];
  int wave = threadIdx.x >> 5, lane = threadIdx.x & 31;
  int jt = blockIdx.x;
  long ncol0 = (long)wave * 512 + jt * 16;   // gate row base in W (N dim)
  v8f acc0 = {}, acc1 = {};
  for (int kb = 0; kb < ga.K; kb += 32) {
    const Seg& sg = ga.s[kb >> 9];
    int kl = kb & 511;
    v16h b  = load_b_frag(sg.W + ncol0 * sg.ws + kl, sg.ws);
    v16h a0 = load_a_frag(sg.A + kl, sg.as);
    v16h a1 = load_a_frag(sg.A + 16 * sg.as + kl, sg.as);
    acc0 = wmma_f16(a0, b, acc0);
    acc1 = wmma_f16(a1, b, acc1);
  }
  int col = lane & 15, hi = lane >> 4;
#pragma unroll
  for (int r = 0; r < 8; ++r) {
    g[wave][r + 8 * hi][col]      = acc0[r];
    g[wave][16 + r + 8 * hi][col] = acc1[r];
  }
  __syncthreads();
#pragma unroll
  for (int q = 0; q < 4; ++q) {
    int idx = threadIdx.x + q * 128;          // 0..511
    int b_ = idx >> 4, c_ = idx & 15;
    int ncol = jt * 16 + c_;
    float gi = g[0][b_][c_], gf = g[1][b_][c_], gg = g[2][b_][c_], go = g[3][b_][c_];
    if (pre) {
      const float* pr = pre + (long)b_ * 2048;
      gi += pr[ncol]; gf += pr[512 + ncol]; gg += pr[1024 + ncol]; go += pr[1536 + ncol];
    }
    if (bias) {
      gi += bias[ncol]; gf += bias[512 + ncol]; gg += bias[1024 + ncol]; go += bias[1536 + ncol];
    }
    float cold = c32[b_ * 512 + ncol];
    float cn = sigm(gf) * cold + sigm(gi) * tanhf(gg);
    float hn = sigm(go) * tanhf(cn);
    c32[b_ * 512 + ncol] = cn;
    h32[b_ * 512 + ncol] = hn;
    h16out[(long)b_ * h16rs + ncol] = (h16)hn;
  }
}

// ---- attention: e = attB.h, softmax over S=48, a = alpha.enc (f16 out) ----
__global__ void k_attn(const float* __restrict__ attB, const float* __restrict__ h32,
                       const h16* __restrict__ enc16, h16* __restrict__ a16) {
  int b = blockIdx.x;
  __shared__ float e[48], alpha[48];
  int wave = threadIdx.x >> 5, lane = threadIdx.x & 31;
  for (int s = wave; s < 48; s += 8) {
    const float* ab = attB + ((long)b * 48 + s) * 512;
    const float* hp = h32 + b * 512;
    float acc = 0.f;
    for (int k = lane; k < 512; k += 32) acc += ab[k] * hp[k];
    for (int off = 16; off >= 1; off >>= 1) acc += __shfl_xor(acc, off, 32);
    if (lane == 0) e[s] = acc;
  }
  __syncthreads();
  if (threadIdx.x == 0) {
    float mx = -3.0e38f;
    for (int s = 0; s < 48; ++s) mx = fmaxf(mx, e[s]);
    float sum = 0.f;
    for (int s = 0; s < 48; ++s) { alpha[s] = __expf(e[s] - mx); sum += alpha[s]; }
    float inv = 1.f / sum;
    for (int s = 0; s < 48; ++s) alpha[s] *= inv;
  }
  __syncthreads();
  for (int d = threadIdx.x; d < 1024; d += 256) {
    float acc = 0.f;
    for (int s = 0; s < 48; ++s)
      acc += alpha[s] * (float)enc16[((long)b * 48 + s) * 1024 + d];
    a16[b * 1024 + d] = (h16)acc;
  }
}

// ---- TDM: stage one 16x512 f16 tile (16KB, contiguous) into LDS ----
// 1-D descriptor per ISA cdna5 §8: count=1, data_size=2B, tile_dim0=8192 elems.
__device__ __forceinline__ void tdm_load_tile_16x512(const h16* gsrc, h16* lds_dst) {
  unsigned lds_off = (unsigned)(unsigned long long)(void*)lds_dst; // LDS aperture: low 32 bits = LDS byte addr
  unsigned long long ga = (unsigned long long)(const void*)gsrc;
  v4u g0;
  g0[0] = 1u;                                           // count=1, is_restore=0, gather=0
  g0[1] = lds_off;                                      // lds_addr
  g0[2] = (unsigned)ga;                                 // global_addr[31:0]
  g0[3] = (unsigned)((ga >> 32) & 0x01FFFFFFu) | (2u << 30); // addr[56:32] | type=2
  v8i g1;
  g1[0] = (int)(1u << 16);       // wg_mask=0, data_size=1 (2 bytes)
  g1[1] = (int)(8192u << 16);    // tensor_dim0[15:0]=8192 in bits 63:48
  g1[2] = (int)(1u << 16);       // tensor_dim0 hi=0 | tensor_dim1[15:0]=1
  g1[3] = (int)(8192u << 16);    // tensor_dim1 hi=0 | tile_dim0=8192
  g1[4] = 1;                     // tile_dim1=1, tile_dim2=0
  g1[5] = 8192;                  // tensor_dim0_stride[31:0]
  g1[6] = (int)(8192u << 16);    // stride0 hi=0 | tensor_dim1_stride[15:0]
  g1[7] = 0;
  v4i g2 = {0, 0, 0, 0}, g3 = {0, 0, 0, 0};
#if defined(__clang_major__) && (__clang_major__ >= 23)
  v8i g4 = {0, 0, 0, 0, 0, 0, 0, 0};
  __builtin_amdgcn_tensor_load_to_lds(g0, g1, g2, g3, g4, 0);
#else
  __builtin_amdgcn_tensor_load_to_lds(g0, g1, g2, g3, 0);
#endif
}

// ---- vocab projection fused with online log-sum-exp + gold gather ----
// grid 96 (M-tiles of Os rows), 256 threads (8 waves over N tiles)
__global__ void k_vocab_nll(const h16* __restrict__ Os16, const h16* __restrict__ Wv16,
                            const int* __restrict__ tgt, float* __restrict__ goldbuf) {
  __shared__ h16 Alds[16 * 512];
  __shared__ float red[8][16][3];
  int mtile = blockIdx.x;
  int wave = threadIdx.x >> 5, lane = threadIdx.x & 31;
  int t = mtile >> 1;
  if (threadIdx.x < 32) {                     // wave 0 issues the TDM copy
    tdm_load_tile_16x512(Os16 + (long)mtile * 16 * 512, Alds);
  }
  __builtin_amdgcn_s_wait_tensorcnt((short)0); // no-op for waves with TENSORcnt==0
  __syncthreads();

  // hoist the whole A tile into registers: 16 K-fragments = 128 VGPRs
  v16h afr[16];
#pragma unroll
  for (int kk = 0; kk < 16; ++kk) afr[kk] = load_a_frag(Alds + kk * 32, 512);

  int col = lane & 15, hi = lane >> 4;
  float rmax[8], rsum[8], rgold[8];
  int gidx[8];
#pragma unroll
  for (int r = 0; r < 8; ++r) {
    rmax[r] = -3.0e38f; rsum[r] = 0.f; rgold[r] = -3.0e38f;
    int b_ = (mtile & 1) * 16 + r + 8 * hi;
    gidx[r] = (t < 47) ? tgt[(t + 1) * 32 + b_] : -1;
  }
  for (int ntile = wave; ntile < 2000; ntile += 8) {
    if (ntile + 8 < 2000)
      __builtin_prefetch((const void*)(Wv16 + (long)(ntile + 8) * 16 * 512), 0, 1);
    v8f acc = {};
#pragma unroll
    for (int kk = 0; kk < 16; ++kk) {
      v16h b = load_b_frag(Wv16 + (long)ntile * 16 * 512 + kk * 32, 512);
      acc = wmma_f16(afr[kk], b, acc);
    }
    int n = ntile * 16 + col;
#pragma unroll
    for (int r = 0; r < 8; ++r) {
      float v = acc[r];
      float mn = fmaxf(rmax[r], v);
      rsum[r] = rsum[r] * __expf(rmax[r] - mn) + __expf(v - mn);
      rmax[r] = mn;
      if (n == gidx[r]) rgold[r] = v;
    }
  }
#pragma unroll
  for (int r = 0; r < 8; ++r) {
    for (int off = 8; off >= 1; off >>= 1) {     // reduce 16 lanes of same row
      float om = __shfl_xor(rmax[r], off, 32);
      float os = __shfl_xor(rsum[r], off, 32);
      float og = __shfl_xor(rgold[r], off, 32);
      float mn = fmaxf(rmax[r], om);
      rsum[r] = rsum[r] * __expf(rmax[r] - mn) + os * __expf(om - mn);
      rmax[r] = mn;
      rgold[r] = fmaxf(rgold[r], og);
    }
    if (col == 0) {
      int m = r + 8 * hi;
      red[wave][m][0] = rmax[r];
      red[wave][m][1] = rsum[r];
      red[wave][m][2] = rgold[r];
    }
  }
  __syncthreads();
  if (threadIdx.x < 16 && t < 47) {
    int m = threadIdx.x;
    float M = -3.0e38f, S = 0.f, G = -3.0e38f;
    for (int w = 0; w < 8; ++w) {
      float om = red[w][m][0], os = red[w][m][1];
      float mn = fmaxf(M, om);
      S = S * __expf(M - mn) + os * __expf(om - mn);
      M = mn;
      G = fmaxf(G, red[w][m][2]);
    }
    float lse = M + __logf(S);
    int b_ = (mtile & 1) * 16 + m;
    float mask = (tgt[(t + 1) * 32 + b_] != 0) ? 1.f : 0.f;
    goldbuf[t * 32 + b_] = (G - lse) * mask;
  }
}

__global__ void k_nll_finalize(const float* __restrict__ goldbuf, float* __restrict__ out) {
  int b = threadIdx.x;
  float s = 0.f;
  for (int t = 0; t < 47; ++t) s += goldbuf[t * 32 + b];
  out[b] = s;
}

// =====================================================================
extern "C" void kernel_launch(void* const* d_in, const int* in_sizes, int n_in,
                              void* d_out, int out_size, void* d_ws, size_t ws_size,
                              hipStream_t stream) {
  (void)in_sizes; (void)n_in; (void)out_size; (void)ws_size;
  // flattened params dict order (only live tensors used; stage 1 is dead code)
  const float* emb_x   = (const float*)d_in[0];
  const float* emb_y   = (const float*)d_in[1];
  const float* f2_Wihf = (const float*)d_in[14];
  const float* f2_Whhf = (const float*)d_in[15];
  const float* f2_bf   = (const float*)d_in[16];
  const float* f2_Wihb = (const float*)d_in[17];
  const float* f2_Whhb = (const float*)d_in[18];
  const float* f2_bb   = (const float*)d_in[19];
  const float* g2_Wihf = (const float*)d_in[20];
  const float* g2_Whhf = (const float*)d_in[21];
  const float* g2_bf   = (const float*)d_in[22];
  const float* g2_Wihb = (const float*)d_in[23];
  const float* g2_Whhb = (const float*)d_in[24];
  const float* g2_bb   = (const float*)d_in[25];
  const float* d_Wih   = (const float*)d_in[33];
  const float* d_Whh   = (const float*)d_in[34];
  const float* d_b     = (const float*)d_in[35];
  const float* d_hproj = (const float*)d_in[36];
  const float* d_cproj = (const float*)d_in[37];
  const float* d_aproj = (const float*)d_in[38];
  const float* d_comb  = (const float*)d_in[39];
  const float* Wvocab  = (const float*)d_in[40];
  const int*   src     = (const int*)d_in[41];
  const int*   tgt     = (const int*)d_in[42];
  float* out = (float*)d_out;

  char* p = (char*)d_ws;
  auto alloc = [&](size_t bytes) -> void* {
    void* r = (void*)p;
    p += (bytes + 255) & ~(size_t)255;
    return r;
  };
  auto a16b = [&](size_t n) -> h16*  { return (h16*)alloc(n * sizeof(h16)); };
  auto a32b = [&](size_t n) -> float* { return (float*)alloc(n * sizeof(float)); };

  // f16 workspace
  h16* ex16 = a16b(1536 * 512);
  h16* ey16 = a16b(1536 * 512);
  h16* Wihf2f = a16b(2048 * 512);
  h16* Whhf2f = a16b(2048 * 512);
  h16* Wihf2b = a16b(2048 * 512);
  h16* Whhf2b = a16b(2048 * 512);
  h16* Wihg2f = a16b(2048 * 512);
  h16* Whhg2f = a16b(2048 * 512);
  h16* Wihg2b = a16b(2048 * 512);
  h16* Whhg2b = a16b(2048 * 512);
  h16* dWih16 = a16b(2048 * 1024);
  h16* dWhh16 = a16b(2048 * 512);
  h16* hproj16 = a16b(512 * 1024);
  h16* cproj16 = a16b(512 * 1024);
  h16* aproj16 = a16b(512 * 1024);
  h16* comb16  = a16b(512 * 1536);
  h16* Wv16    = a16b((size_t)32000 * 512);
  h16* enc16   = a16b((size_t)32 * 48 * 1024);   // [b][s][1024]
  h16* g2hf[2] = { a16b(32 * 512), a16b(32 * 512) };
  h16* g2hb[2] = { a16b(32 * 512), a16b(32 * 512) };
  h16* hycat16 = a16b(32 * 1024);
  h16* cycat16 = a16b(32 * 1024);
  h16* dh[2]   = { a16b(32 * 512), a16b(32 * 512) };
  h16* a16     = a16b(32 * 1024);
  h16* Os16    = a16b(1536 * 512);
  h16* zeros16 = a16b(32 * 512);
  // f32 workspace
  float* xgf  = a32b((size_t)1536 * 2048);
  float* xgb  = a32b((size_t)1536 * 2048);
  float* ygf  = a32b((size_t)1536 * 2048);
  float* ygb  = a32b((size_t)1536 * 2048);
  float* attB = a32b((size_t)32 * 48 * 512);
  float* hf2f = a32b(32 * 512); float* cf2f = a32b(32 * 512);
  float* hf2b = a32b(32 * 512); float* cf2b = a32b(32 * 512);
  float* hg2f = a32b(32 * 512); float* cg2f = a32b(32 * 512);
  float* hg2b = a32b(32 * 512); float* cg2b = a32b(32 * 512);
  float* hdec = a32b(32 * 512); float* cdec = a32b(32 * 512);
  float* goldbuf = a32b(47 * 32);

  // zero initial states
  hipMemsetAsync(cf2f, 0, 32 * 512 * 4, stream);
  hipMemsetAsync(cf2b, 0, 32 * 512 * 4, stream);
  hipMemsetAsync(cg2f, 0, 32 * 512 * 4, stream);
  hipMemsetAsync(cg2b, 0, 32 * 512 * 4, stream);
  hipMemsetAsync(zeros16, 0, 32 * 512 * sizeof(h16), stream);

  auto conv = [&](const float* s, h16* d, long n) {
    int g = (int)((n + 255) / 256); if (g > 4096) g = 4096;
    k_f32_to_f16<<<g, 256, 0, stream>>>(s, d, n);
  };
  conv(f2_Wihf, Wihf2f, 2048 * 512); conv(f2_Whhf, Whhf2f, 2048 * 512);
  conv(f2_Wihb, Wihf2b, 2048 * 512); conv(f2_Whhb, Whhf2b, 2048 * 512);
  conv(g2_Wihf, Wihg2f, 2048 * 512); conv(g2_Whhf, Whhg2f, 2048 * 512);
  conv(g2_Wihb, Wihg2b, 2048 * 512); conv(g2_Whhb, Whhg2b, 2048 * 512);
  conv(d_Wih, dWih16, 2048 * 1024);  conv(d_Whh, dWhh16, 2048 * 512);
  conv(d_hproj, hproj16, 512 * 1024); conv(d_cproj, cproj16, 512 * 1024);
  conv(d_aproj, aproj16, 512 * 1024); conv(d_comb, comb16, 512 * 1536);
  conv(Wvocab, Wv16, (long)32000 * 512);

  k_gather16<<<1536, 256, 0, stream>>>(emb_x, src, ex16);
  k_gather16<<<1536, 256, 0, stream>>>(emb_y, tgt, ey16);

  // input projections xg = x @ Wih^T + b   (M=1536, N=2048, K=512)
  {
    GemmArgs ga = {}; ga.nseg = 1; ga.K = 512;
    ga.s[0] = { ex16, 512, Wihf2f, 512 };
    k_gemm<<<dim3(96, 32), 32, 0, stream>>>(ga, f2_bf, xgf, nullptr, 2048, 0);
    ga.s[0] = { ex16, 512, Wihf2b, 512 };
    k_gemm<<<dim3(96, 32), 32, 0, stream>>>(ga, f2_bb, xgb, nullptr, 2048, 0);
    ga.s[0] = { ey16, 512, Wihg2f, 512 };
    k_gemm<<<dim3(96, 32), 32, 0, stream>>>(ga, g2_bf, ygf, nullptr, 2048, 0);
    ga.s[0] = { ey16, 512, Wihg2b, 512 };
    k_gemm<<<dim3(96, 32), 32, 0, stream>>>(ga, g2_bb, ygb, nullptr, 2048, 0);
  }

  // encoder f2 forward (h_t -> enc16[b][t][0:512])
  for (int t = 0; t < 48; ++t) {
    GemmArgs sa = {}; sa.nseg = 1; sa.K = 512;
    if (t == 0) sa.s[0] = { zeros16, 512, Whhf2f, 512 };
    else        sa.s[0] = { enc16 + (long)(t - 1) * 1024, 48 * 1024, Whhf2f, 512 };
    k_lstm_step<<<32, 128, 0, stream>>>(sa, xgf + (long)t * 32 * 2048, nullptr,
                                        hf2f, cf2f, enc16 + (long)t * 1024, 48 * 1024);
  }
  // encoder f2 backward (h_t -> enc16[b][t][512:1024])
  for (int t = 47; t >= 0; --t) {
    GemmArgs sa = {}; sa.nseg = 1; sa.K = 512;
    if (t == 47) sa.s[0] = { zeros16, 512, Whhf2b, 512 };
    else         sa.s[0] = { enc16 + (long)(t + 1) * 1024 + 512, 48 * 1024, Whhf2b, 512 };
    k_lstm_step<<<32, 128, 0, stream>>>(sa, xgb + (long)t * 32 * 2048, nullptr,
                                        hf2b, cf2b, enc16 + (long)t * 1024 + 512, 48 * 1024);
  }
  // encoder g2 forward / backward (only final states matter)
  for (int t = 0; t < 48; ++t) {
    GemmArgs sa = {}; sa.nseg = 1; sa.K = 512;
    sa.s[0] = { (t == 0) ? zeros16 : g2hf[(t & 1) ^ 1], 512, Whhg2f, 512 };
    k_lstm_step<<<32, 128, 0, stream>>>(sa, ygf + (long)t * 32 * 2048, nullptr,
                                        hg2f, cg2f, g2hf[t & 1], 512);
  }
  for (int i = 0; i < 48; ++i) {
    int t = 47 - i;
    GemmArgs sa = {}; sa.nseg = 1; sa.K = 512;
    sa.s[0] = { (i == 0) ? zeros16 : g2hb[(i & 1) ^ 1], 512, Whhg2b, 512 };
    k_lstm_step<<<32, 128, 0, stream>>>(sa, ygb + (long)t * 32 * 2048, nullptr,
                                        hg2b, cg2b, g2hb[i & 1], 512);
  }
  k_pack16<<<64, 256, 0, stream>>>(hg2f, hycat16, 1024, 0);
  k_pack16<<<64, 256, 0, stream>>>(hg2b, hycat16, 1024, 512);
  k_pack16<<<64, 256, 0, stream>>>(cg2f, cycat16, 1024, 0);
  k_pack16<<<64, 256, 0, stream>>>(cg2b, cycat16, 1024, 512);

  // attB = enc @ att_proj^T  (M=1536 rows b*48+s, N=512, K=1024)
  {
    GemmArgs ga = {}; ga.nseg = 2; ga.K = 1024;
    ga.s[0] = { enc16, 1024, aproj16, 1024 };
    ga.s[1] = { enc16 + 512, 1024, aproj16 + 512, 1024 };
    k_gemm<<<dim3(96, 8), 32, 0, stream>>>(ga, nullptr, attB, nullptr, 512, 0);
  }
  // h0 / c0 projections (M=32, N=512, K=1024)
  {
    GemmArgs ga = {}; ga.nseg = 2; ga.K = 1024;
    ga.s[0] = { hycat16, 1024, hproj16, 1024 };
    ga.s[1] = { hycat16 + 512, 1024, hproj16 + 512, 1024 };
    k_gemm<<<dim3(2, 8), 32, 0, stream>>>(ga, nullptr, hdec, dh[1], 512, 0);
    ga.s[0] = { cycat16, 1024, cproj16, 1024 };
    ga.s[1] = { cycat16 + 512, 1024, cproj16 + 512, 1024 };
    k_gemm<<<dim3(2, 8), 32, 0, stream>>>(ga, nullptr, cdec, nullptr, 512, 0);
  }

  // attention decoder
  for (int t = 0; t < 48; ++t) {
    GemmArgs sa = {}; sa.nseg = 3; sa.K = 1536;
    sa.s[0] = { ey16 + (long)t * 32 * 512, 512, dWih16, 1024 };        // y_t
    sa.s[1] = { (t == 0) ? zeros16 : Os16 + (long)(t - 1) * 32 * 512,
                512, dWih16 + 512, 1024 };                             // o_prev
    sa.s[2] = { dh[(t + 1) & 1], 512, dWhh16, 512 };                   // h_prev
    k_lstm_step<<<32, 128, 0, stream>>>(sa, nullptr, d_b, hdec, cdec,
                                        dh[t & 1], 512);
    k_attn<<<32, 256, 0, stream>>>(attB, hdec, enc16, a16);
    GemmArgs gc = {}; gc.nseg = 3; gc.K = 1536;
    gc.s[0] = { a16, 1024, comb16, 1536 };
    gc.s[1] = { a16 + 512, 1024, comb16 + 512, 1536 };
    gc.s[2] = { dh[t & 1], 512, comb16 + 1024, 1536 };
    k_gemm<<<dim3(2, 8), 32, 0, stream>>>(gc, nullptr, nullptr,
                                          Os16 + (long)t * 32 * 512, 512, 1);
  }

  // vocab projection + streaming log-softmax + gold gather + batch sum
  k_vocab_nll<<<96, 256, 0, stream>>>(Os16, Wv16, tgt, goldbuf);
  k_nll_finalize<<<1, 32, 0, stream>>>(goldbuf, out);
}
